// NonlinearFilter_82051055223038
// MI455X (gfx1250) — compile-verified, matrix operand order checked by reading, not run on hardware
//
#include <hip/hip_runtime.h>
#include <hip/hip_bf16.h>

// ---------------------------------------------------------------------------
// NonlinearFilter on MI455X (gfx1250)
//
//  K0: convert W1/W2 -> bf16 (workspace, L2-resident for the scan)
//  K1 (N*T=16384 WGs, parallel): M=K^T diag(Q) K, Cholesky L (packed -> ws),
//     klbase = 0.5*(||L^-1||_F^2 - R + logdet)  [uses M = LL^T - I identity],
//     z_p (output), v, chol_solve, z_pmc = z_p - K*cs_v staged into the z_f
//     output slot. Handles all of t=0 (state-independent).
//  K2 (64 WGs, one per n, 8 waves): sequential scan t=1..T-1. bf16 WMMA for
//     the MLP + K-einsums (fp32 accum), wave-parallel fp32 triangular solves.
//     K_t (16KB) and packed L (2.1KB) for step t+1 are DMA'd into LDS by the
//     Tensor Data Mover issued right after their last use in step t, so the
//     DMA overlaps the end-of-step reductions (TENSORcnt wait
//     only on wave 0 at the loop top).
// ---------------------------------------------------------------------------

typedef __attribute__((ext_vector_type(16))) __bf16 bf16x16;
typedef __attribute__((ext_vector_type(8)))  float  f32x8;

constexpr int S = 16, N = 64, T = 256, D = 128, R = 32, H = 512;
constexpr int LPACK = 528;  // 32*33/2 packed lower-triangular floats

// ---- TDM support (arity differs across toolchains) ------------------------
#if defined(__HIP_DEVICE_COMPILE__) && defined(__has_builtin)
#if __has_builtin(__builtin_amdgcn_tensor_load_to_lds) && \
    __has_builtin(__builtin_amdgcn_s_wait_tensorcnt)
#define HAVE_TDM 1
#endif
#endif
#ifndef HAVE_TDM
#define HAVE_TDM 0
#endif

#if HAVE_TDM
typedef unsigned int u32x4 __attribute__((ext_vector_type(4)));
typedef int          i32x4 __attribute__((ext_vector_type(4)));
typedef int          i32x8 __attribute__((ext_vector_type(8)));

// 1-D tile (nelems 4-byte elements) global -> LDS via the Tensor Data Mover.
// D# per ISA 8.3/8.4: group0 = {count, lds_addr, global_addr, type=2},
// group1 = {data_size=4B, tensor_dim0=nelems, tensor_dim1=1, tile_dim0=nelems}.
__device__ __forceinline__ void tdm_load_1d(const void* gptr, void* lptr,
                                            int nelems) {
  unsigned long long ga = (unsigned long long)gptr;
  unsigned int lds = (unsigned int)(unsigned long long)lptr;  // low 32 bits = LDS byte addr
  u32x4 g0;
  g0[0] = 1u;                                  // count=1 (user descriptor)
  g0[1] = lds;                                 // lds_addr
  g0[2] = (unsigned int)(ga & 0xFFFFFFFFu);    // global_addr[31:0]
  g0[3] = (unsigned int)((ga >> 32) & 0x01FFFFFFu) | (2u << 30);  // [56:32] | type=2
  i32x8 g1;
  g1[0] = (int)(2u << 16);                     // data_size = 2 (4 bytes)
  g1[1] = (int)((unsigned)(nelems & 0xFFFF) << 16);   // tensor_dim0[15:0] @bits63:48
  g1[2] = (int)(((unsigned)nelems >> 16) | (1u << 16)); // tensor_dim0[31:16], tensor_dim1=1
  g1[3] = (int)((unsigned)(nelems & 0xFFFF) << 16);   // tile_dim0 @bits127:112
  g1[4] = 0;                                   // tile_dim1/2 unused
  g1[5] = nelems;                              // tensor_dim0_stride[31:0]
  g1[6] = 0;
  g1[7] = 0;
  i32x4 z4 = {0, 0, 0, 0};
#if __clang_major__ >= 23
  i32x8 z8 = {0, 0, 0, 0, 0, 0, 0, 0};
  __builtin_amdgcn_tensor_load_to_lds(g0, g1, z4, z4, z8, 0);
#else
  __builtin_amdgcn_tensor_load_to_lds(g0, g1, z4, z4, 0);
#endif
}
#endif  // HAVE_TDM

// ---- WMMA helpers ----------------------------------------------------------
__device__ __forceinline__ f32x8 wmma_bf16(bf16x16 a, bf16x16 b, f32x8 c) {
  return __builtin_amdgcn_wmma_f32_16x16x32_bf16(false, a, false, b, (short)0, c,
                                                 false, false);
}

// 16-bit A-matrix K mapping (ISA 7.12.2)
__device__ __forceinline__ int kmap(int e, int half) {
  return ((e >> 3) << 4) + (half << 3) + (e & 7);
}

__device__ __forceinline__ bf16x16 fragA_f32(const float* src, int m, int ld,
                                             int kbase, int half) {
  bf16x16 f;
#pragma unroll
  for (int e = 0; e < 16; ++e) f[e] = (__bf16)src[m * ld + kbase + kmap(e, half)];
  return f;
}
__device__ __forceinline__ bf16x16 fragA_bf16(const __bf16* src, int m, int ld,
                                              int kbase, int half) {
  bf16x16 f;
#pragma unroll
  for (int e = 0; e < 16; ++e) f[e] = src[m * ld + kbase + kmap(e, half)];
  return f;
}
__device__ __forceinline__ bf16x16 fragB(const __bf16* src, int col, int ld,
                                         int kbase, int half) {
  bf16x16 f;
#pragma unroll
  for (int e = 0; e < 16; ++e) f[e] = src[(kbase + kmap(e, half)) * ld + col];
  return f;
}
__device__ __forceinline__ bf16x16 fragB_f32(const float* src, int col, int ld,
                                             int kbase, int half) {
  bf16x16 f;
#pragma unroll
  for (int e = 0; e < 16; ++e)
    f[e] = (__bf16)src[(kbase + kmap(e, half)) * ld + col];
  return f;
}

// Wave-parallel Cholesky of 32x32 (I+M) in LDS; writes packed L and 1/diag.
__device__ __forceinline__ void cholesky32(float* Mm, float* Lp, float* rdiag,
                                           int lane) {
  for (int j = 0; j < 32; ++j) {
    float djj = sqrtf(Mm[j * 32 + j]);
    float inv = 1.0f / djj;
    float lpj = (lane > j) ? Mm[lane * 32 + j] * inv : 0.0f;
    if (lane == j) { Lp[j * (j + 1) / 2 + j] = djj; rdiag[j] = inv; }
    if (lane > j)  Lp[lane * (lane + 1) / 2 + j] = lpj;
    for (int q = j + 1; q < 32; ++q) {
      float lqj = __shfl(lpj, q, 32);
      if (lane >= q) Mm[lane * 32 + q] -= lpj * lqj;
    }
  }
}

// ||L^-1||_F^2 via 32 column forward-solves (lane = column, X scratch 32x32)
__device__ __forceinline__ float linv_fro2(const float* Lp, const float* rdiag,
                                           float* X, int lane) {
  float acc = 0.0f;
  for (int i = 0; i < 32; ++i) {
    float s = (i == lane) ? 1.0f : 0.0f;
    for (int p = lane; p < i; ++p) s -= Lp[i * (i + 1) / 2 + p] * X[p * 32 + lane];
    s *= rdiag[i];
    X[i * 32 + lane] = s;
    acc += s * s;
  }
  for (int off = 16; off; off >>= 1) acc += __shfl_down(acc, off, 32);
  return acc;  // valid on lane 0
}

// Solve (L L^T) x = y for one row; lane holds component p.
__device__ __forceinline__ float chol_solve_row(float y, const float* Lp,
                                                const float* rdiag, int lane) {
  for (int i = 0; i < 32; ++i) {  // forward: L z = y
    float yi = __shfl(y, i, 32) * rdiag[i];
    if (lane == i) y = yi;
    else if (lane > i) y -= Lp[lane * (lane + 1) / 2 + i] * yi;
  }
  for (int i = 31; i >= 0; --i) {  // backward: L^T x = z
    float xi = __shfl(y, i, 32) * rdiag[i];
    if (lane == i) y = xi;
    else if (lane < i) y -= Lp[i * (i + 1) / 2 + lane] * xi;
  }
  return y;
}

// ---------------------------------------------------------------------------
__global__ __launch_bounds__(256) void convert_weights(const float* __restrict__ W1,
                                                       const float* __restrict__ W2,
                                                       __bf16* __restrict__ W1b,
                                                       __bf16* __restrict__ W2b) {
  int i = blockIdx.x * 256 + threadIdx.x;
  if (i < D * H) W1b[i] = (__bf16)W1[i];
  if (i < H * D) W2b[i] = (__bf16)W2[i];
}

// ---------------------------------------------------------------------------
__global__ __launch_bounds__(256) void precomp_kernel(
    const float* __restrict__ k_in, const float* __restrict__ K_in,
    const float* __restrict__ eps_w, const float* __restrict__ eps_z,
    const float* __restrict__ Qd, const float* __restrict__ Q0d,
    const float* __restrict__ m_init, float* __restrict__ Lpack,
    float* __restrict__ klbase, float* __restrict__ out_zf,
    float* __restrict__ out_mf, float* __restrict__ out_zp,
    float* __restrict__ out_kl) {
  __shared__ float Kf[D * R];        // 16KB fp32 K_t (TDM destination)
  __shared__ float Mm[R * R];        // 4KB (M, then X scratch, then t=0 stash)
  __shared__ float Lp1[LPACK];
  __shared__ float rdiag1[R];
  __shared__ float Qs1[D], sqQ[D];
  __shared__ float zp[S * D];        // 8KB
  __shared__ float stag1[8 * 256];   // 8KB K-split partials / z_pmc stash
  __shared__ float vbuf[S * R];      // 2KB
  __shared__ float cs1[S * R];       // 2KB
  __shared__ float x0[D];
  __shared__ float klbsh;

  const int tid = threadIdx.x, lane = tid & 31, w = tid >> 5;
  const int n = blockIdx.x / T, t = blockIdx.x % T;
  const float* Kt = K_in + ((size_t)n * T + t) * D * R;

#if HAVE_TDM
  if (w == 0) tdm_load_1d(Kt, Kf, D * R);
#else
  for (int i = tid; i < D * R; i += 256) Kf[i] = Kt[i];
#endif
  for (int i = tid; i < D; i += 256) {
    float q = (t == 0) ? Q0d[i] : Qd[i];
    Qs1[i] = q; sqQ[i] = sqrtf(q);
  }
#if HAVE_TDM
  if (w == 0) __builtin_amdgcn_s_wait_tensorcnt((short)0);
#endif
  __syncthreads();

  // M = K^T diag(Q_chol) K + I
  for (int e = tid; e < R * R; e += 256) {
    int r = e >> 5, c = e & 31;
    float acc = (r == c) ? 1.0f : 0.0f;
    for (int d = 0; d < D; ++d) acc += Qs1[d] * Kf[d * R + r] * Kf[d * R + c];
    Mm[e] = acc;
  }
  __syncthreads();

  if (w == 0) {  // Cholesky + KL base on wave 0; other waves compute z_p
    cholesky32(Mm, Lp1, rdiag1, lane);
    float fro2 = linv_fro2(Lp1, rdiag1, Mm, lane);
    float ld = -logf(rdiag1[lane]);
    for (int off = 16; off; off >>= 1) ld += __shfl_down(ld, off, 32);
    if (lane == 0) {
      float kb = 0.5f * ((fro2 - (float)R) + 2.0f * ld);  // 0.5*(tr + logdet)
      klbase[n * T + t] = kb; klbsh = kb;
    }
  } else {
    for (int i = tid - 32; i < S * D; i += 224) {
      int s2 = i >> 7, d = i & 127;
      float v = sqQ[d] * eps_z[(((size_t)t * S + s2) * N + n) * D + d];
      zp[i] = v;
      out_zp[(((size_t)s2 * N + n) * T + t) * D + d] = v;
    }
  }
  __syncthreads();

  // v = z_p @ K (16x128 @ 128x32), K-split over 8 waves -> WMMA
  {
    int col = lane & 15, half = lane >> 4, nt = w & 1, kb = w >> 1;
    f32x8 acc = {0, 0, 0, 0, 0, 0, 0, 0};
    bf16x16 a = fragA_f32(zp, col, D, kb * 32, half);
    bf16x16 b = fragB_f32(Kf, nt * 16 + col, R, kb * 32, half);
    acc = wmma_bf16(a, b, acc);
#pragma unroll
    for (int i = 0; i < 8; ++i) stag1[w * 256 + (i + 8 * half) * 16 + col] = acc[i];
  }
  __syncthreads();
  for (int i = tid; i < LPACK; i += 256)
    Lpack[((size_t)n * T + t) * LPACK + i] = Lp1[i];
  for (int e = tid; e < S * R; e += 256) {
    int m = e >> 5, r = e & 31, nt = r >> 4, col = r & 15;
    float s = stag1[nt * 256 + m * 16 + col] + stag1[(2 | nt) * 256 + m * 16 + col] +
              stag1[(4 | nt) * 256 + m * 16 + col] + stag1[(6 | nt) * 256 + m * 16 + col];
    vbuf[e] = s + eps_w[(((size_t)t * S + m) * N + n) * R + r];
  }
  __syncthreads();
  for (int rr = 0; rr < 2; ++rr) {  // per-row (LL^T) solves, wave-parallel
    int s2 = w + rr * 8;
    float y = vbuf[s2 * 32 + lane];
    y = chol_solve_row(y, Lp1, rdiag1, lane);
    cs1[s2 * 32 + lane] = y;
  }
  __syncthreads();
  {  // z_pmc = z_p - cs_v @ K^T  (16x32 @ 32x128)
    int col = lane & 15, half = lane >> 4, dt = w;
    f32x8 acc = {0, 0, 0, 0, 0, 0, 0, 0};
    bf16x16 a = fragA_f32(cs1, col, R, 0, half);
    bf16x16 b;
#pragma unroll
    for (int e = 0; e < 16; ++e)
      b[e] = (__bf16)Kf[(dt * 16 + col) * R + kmap(e, half)];
    acc = wmma_bf16(a, b, acc);
    int d = dt * 16 + col;
#pragma unroll
    for (int i = 0; i < 8; ++i) {
      int m = i + 8 * half;
      float zpmc = zp[m * D + d] - acc[i];
      stag1[m * D + d] = zpmc;
      if (t > 0) out_zf[(((size_t)m * N + n) * T + t) * D + d] = zpmc;
    }
  }

  if (t == 0) {  // full t=0 step (state-independent: m_pred = m_init)
    __syncthreads();
    for (int i = tid; i < D; i += 256)
      x0[i] = m_init[i] + Qd[i] * k_in[((size_t)n * T) * D + i];  // Q_inner = Q_diag
    __syncthreads();
    if (w == 0) {
      float u = 0.0f;
      for (int d = 0; d < D; ++d) u += Kf[d * R + lane] * x0[d];
      u = chol_solve_row(u, Lp1, rdiag1, lane);
      vbuf[lane] = u;
    }
    __syncthreads();
    for (int d = tid; d < D; d += 256) {
      float corr = 0.0f;
      for (int r = 0; r < R; ++r) corr += Kf[d * R + r] * vbuf[r];
      float g = k_in[((size_t)n * T) * D + d] - corr;
      float mf = m_init[d] + Qs1[d] * g;  // Q_chol = Q0 at t=0
      out_mf[((size_t)n * T) * D + d] = mf;
      Mm[d] = Qs1[d] * g * g;
      Mm[128 + d] = mf;
    }
    __syncthreads();
    for (int i = tid; i < S * D; i += 256) {
      int s2 = i >> 7, d = i & 127;
      out_zf[(((size_t)s2 * N + n) * T) * D + d] = stag1[i] + Mm[128 + d];
    }
    if (tid == 0) {
      float qp = 0.0f;
      for (int d2 = 0; d2 < D; ++d2) qp += Mm[d2];
      out_kl[(size_t)n * T] = klbsh + 0.5f * qp;
    }
  }
}

// ---------------------------------------------------------------------------
union UScan {
  __bf16 hbuf[S * H];  // 16KB (live: G1 -> G2)
  struct {             // live after G2
    float stag[8 * 256];  // K-split partials, then g buffer
    float ubuf[S * R];
    float csbuf[S * R];
  } s;
};

__global__ __launch_bounds__(256) void scan_kernel(
    const float* __restrict__ k_in, const float* __restrict__ K_in,
    const float* __restrict__ Qd, const float* __restrict__ b1,
    const float* __restrict__ b2, const __bf16* __restrict__ W1b,
    const __bf16* __restrict__ W2b, const float* __restrict__ Lpack,
    const float* __restrict__ klbase, float* __restrict__ out_zf,
    float* __restrict__ out_mf, float* __restrict__ out_kl) {
  __shared__ float  zprev[S * D];   // 8KB
  __shared__ float  mpred[S * D];   // 8KB
  __shared__ __bf16 xb[S * D];      // 4KB
  __shared__ float  Kf[D * R];      // 16KB fp32 K_t (TDM destination)
  __shared__ float  Lp[LPACK];      // TDM destination
  __shared__ float  rdiag[R];
  __shared__ float  kts[D];
  __shared__ float  b1s[H];
  __shared__ float  b2s[D];
  __shared__ float  Qs[D];
  __shared__ float  qp_part[256];
  __shared__ UScan  us;

  const int tid = threadIdx.x, lane = tid & 31, w = tid >> 5;
  const int n = blockIdx.x;
  const int col = lane & 15, half = lane >> 4;

#if HAVE_TDM
  // pipeline prologue: DMA step-1 K_t and L while we stage constants
  if (w == 0) {
    tdm_load_1d(K_in + ((size_t)n * T + 1) * D * R, Kf, D * R);
    tdm_load_1d(Lpack + ((size_t)n * T + 1) * LPACK, Lp, LPACK);
  }
#endif
  for (int i = tid; i < H; i += 256) b1s[i] = b1[i];
  for (int i = tid; i < D; i += 256) { b2s[i] = b2[i]; Qs[i] = Qd[i]; }
  for (int i = tid; i < S * D; i += 256) {  // z_prev = z_f0 (written by precomp)
    int s2 = i >> 7, d = i & 127;
    zprev[i] = out_zf[(((size_t)s2 * N + n) * T) * D + d];
  }
  __syncthreads();

  for (int t = 1; t < T; ++t) {
    for (int i = tid; i < D; i += 256) kts[i] = k_in[((size_t)n * T + t) * D + i];
#if HAVE_TDM
    if (w == 0) __builtin_amdgcn_s_wait_tensorcnt((short)0);
#else
    {
      const float* Kt = K_in + ((size_t)n * T + t) * D * R;
      for (int i = tid; i < D * R; i += 256) Kf[i] = Kt[i];
      for (int i = tid; i < LPACK; i += 256)
        Lp[i] = Lpack[((size_t)n * T + t) * LPACK + i];
    }
#endif
    if (t + 1 < T && tid < 8)  // warm L2 for next k_t (global_prefetch_b8)
      __builtin_prefetch(k_in + ((size_t)n * T + t + 1) * D + tid * 16, 0, 0);
    __syncthreads();
    if (tid < 32) rdiag[tid] = 1.0f / Lp[tid * (tid + 1) / 2 + tid];

    // G1: h = tanh(z W1 + b1)  (16x128 @ 128x512), 4 N-tiles per wave
#pragma unroll
    for (int j = 0; j < 4; ++j) {
      int nt = w * 4 + j;
      f32x8 acc = {0, 0, 0, 0, 0, 0, 0, 0};
#pragma unroll
      for (int kb = 0; kb < 4; ++kb) {
        bf16x16 a = fragA_f32(zprev, col, D, kb * 32, half);
        bf16x16 b = fragB(W1b, nt * 16 + col, H, kb * 32, half);
        acc = wmma_bf16(a, b, acc);
      }
      float bb = b1s[nt * 16 + col];
#pragma unroll
      for (int i = 0; i < 8; ++i)
        us.hbuf[(i + 8 * half) * H + nt * 16 + col] = (__bf16)tanhf(acc[i] + bb);
    }
    __syncthreads();

    // G2: m_pred = z + h W2 + b2  (16x512 @ 512x128), 1 N-tile per wave
    {
      int nt = w;
      f32x8 acc = {0, 0, 0, 0, 0, 0, 0, 0};
#pragma unroll 4
      for (int kb = 0; kb < 16; ++kb) {
        bf16x16 a = fragA_bf16(us.hbuf, col, H, kb * 32, half);
        bf16x16 b = fragB(W2b, nt * 16 + col, D, kb * 32, half);
        acc = wmma_bf16(a, b, acc);
      }
      int d = nt * 16 + col;
      float b2v = b2s[d], qv = Qs[d], kv = kts[d];
#pragma unroll
      for (int i = 0; i < 8; ++i) {
        int m = i + 8 * half;
        float mp = zprev[m * D + d] + acc[i] + b2v;
        mpred[m * D + d] = mp;
        xb[m * D + d] = (__bf16)(mp + qv * kv);  // x = m_pred + Q*k_t
      }
    }
    __syncthreads();

    // G4: u = x @ K  (16x128 @ 128x32), K-split over waves + LDS reduce
    {
      int nt = w & 1, kb = w >> 1;
      f32x8 acc = {0, 0, 0, 0, 0, 0, 0, 0};
      bf16x16 a = fragA_bf16(xb, col, D, kb * 32, half);
      bf16x16 b = fragB_f32(Kf, nt * 16 + col, R, kb * 32, half);
      acc = wmma_bf16(a, b, acc);
#pragma unroll
      for (int i = 0; i < 8; ++i)
        us.s.stag[w * 256 + (i + 8 * half) * 16 + col] = acc[i];
    }
    __syncthreads();
    for (int e = tid; e < S * R; e += 256) {
      int m = e >> 5, r = e & 31, nt = r >> 4, c2 = r & 15;
      us.s.ubuf[e] = us.s.stag[nt * 256 + m * 16 + c2] +
                     us.s.stag[(2 | nt) * 256 + m * 16 + c2] +
                     us.s.stag[(4 | nt) * 256 + m * 16 + c2] +
                     us.s.stag[(6 | nt) * 256 + m * 16 + c2];
    }
    __syncthreads();

    // per-row chol solves (2 rows per wave)
    for (int rr = 0; rr < 2; ++rr) {
      int s2 = w + rr * 8;
      float y = us.s.ubuf[s2 * 32 + lane];
      y = chol_solve_row(y, Lp, rdiag, lane);
      us.s.csbuf[s2 * 32 + lane] = y;
    }
    __syncthreads();

    // G5: corr = cs_u @ K^T; epilogue: g, m_f, z_f
    {
      int dt = w;
      f32x8 acc = {0, 0, 0, 0, 0, 0, 0, 0};
      bf16x16 a = fragA_f32(us.s.csbuf, col, R, 0, half);
      bf16x16 b;
#pragma unroll
      for (int e = 0; e < 16; ++e)
        b[e] = (__bf16)Kf[(dt * 16 + col) * R + kmap(e, half)];
      acc = wmma_bf16(a, b, acc);
      int d = dt * 16 + col;
      float kv = kts[d], qv = Qs[d];
#pragma unroll
      for (int i = 0; i < 8; ++i) {
        int m = i + 8 * half;
        float g = kv - acc[i];
        float mf = mpred[m * D + d] + qv * g;
        size_t oz = (((size_t)m * N + n) * T + t) * D + d;
        float zf = mf + out_zf[oz];  // z_pmc staged by precomp
        out_zf[oz] = zf;
        zprev[m * D + d] = zf;
        us.s.stag[m * D + d] = g;  // g buffer (stag is dead)
      }
    }
    __syncthreads();

#if HAVE_TDM
    // Kf/Lp are dead now: DMA next step's tiles; overlaps the reductions below
    if (w == 0 && t + 1 < T) {
      tdm_load_1d(K_in + ((size_t)n * T + t + 1) * D * R, Kf, D * R);
      tdm_load_1d(Lpack + ((size_t)n * T + t + 1) * LPACK, Lp, LPACK);
    }
#endif

    // reductions: m_f mean over s, kl = klbase + 0.5*mean_s(sum Q g^2)
    if (tid < D) {
      float acc = 0.0f;
      for (int s2 = 0; s2 < S; ++s2)
        acc += mpred[s2 * D + tid] + Qs[tid] * us.s.stag[s2 * D + tid];
      out_mf[((size_t)n * T + t) * D + tid] = acc * (1.0f / S);
    }
    {
      int s2 = tid >> 4, j = tid & 15;
      float p = 0.0f;
      for (int dd = 0; dd < 8; ++dd) {
        int d = j * 8 + dd;
        float g = us.s.stag[s2 * D + d];
        p += Qs[d] * g * g;
      }
      qp_part[tid] = p;
    }
    __syncthreads();
    if (tid == 0) {
      float q = 0.0f;
      for (int i2 = 0; i2 < 256; ++i2) q += qp_part[i2];
      out_kl[(size_t)n * T + t] = klbase[(size_t)n * T + t] + 0.5f * q * (1.0f / S);
    }
    __syncthreads();
  }
}

// ---------------------------------------------------------------------------
extern "C" void kernel_launch(void* const* d_in, const int* in_sizes, int n_in,
                              void* d_out, int out_size, void* d_ws, size_t ws_size,
                              hipStream_t stream) {
  (void)in_sizes; (void)n_in; (void)out_size; (void)ws_size;
  const float* k    = (const float*)d_in[0];
  const float* Kt   = (const float*)d_in[1];
  const float* ew   = (const float*)d_in[2];
  const float* ez   = (const float*)d_in[3];
  const float* W1   = (const float*)d_in[4];
  const float* b1   = (const float*)d_in[5];
  const float* W2   = (const float*)d_in[6];
  const float* b2   = (const float*)d_in[7];
  const float* Qd   = (const float*)d_in[8];
  const float* Q0   = (const float*)d_in[9];
  const float* mi   = (const float*)d_in[10];

  float* out    = (float*)d_out;
  float* out_zf = out;                                   // (S,N,T,D)
  float* out_mf = out + (size_t)S * N * T * D;           // (N,T,D)
  float* out_zp = out_mf + (size_t)N * T * D;            // (S,N,T,D)
  float* out_kl = out_zp + (size_t)S * N * T * D;        // (N,T)

  // workspace: W1b | W2b | klbase | Lpack  (~35 MB total)
  char* ws = (char*)d_ws;
  __bf16* W1b = (__bf16*)ws;
  __bf16* W2b = (__bf16*)(ws + 131072);
  float*  klb = (float*)(ws + 262144);
  float*  Lpk = (float*)(ws + 262144 + 65536);

  convert_weights<<<256, 256, 0, stream>>>(W1, W2, W1b, W2b);
  precomp_kernel<<<N * T, 256, 0, stream>>>(k, Kt, ew, ez, Qd, Q0, mi, Lpk, klb,
                                            out_zf, out_mf, out_zp, out_kl);
  scan_kernel<<<N, 256, 0, stream>>>(k, Kt, Qd, b1, b2, W1b, W2b, Lpk, klb,
                                     out_zf, out_mf, out_kl);
}
